// GraphAutoEncoder_36885179138300
// MI455X (gfx1250) — matrix operand order repaired and checked by reading, not compiled, not run on hardware
//
#include <hip/hip_runtime.h>
#include <hip/hip_bf16.h>

// ---------------------------------------------------------------------------
// Types for CDNA5 WMMA
// ---------------------------------------------------------------------------
typedef __attribute__((ext_vector_type(16))) __bf16        v16bf;
typedef __attribute__((ext_vector_type(8)))  float         v8f;
typedef __attribute__((ext_vector_type(8)))  unsigned int  v8u;
typedef __attribute__((ext_vector_type(4)))  unsigned int  v4u;
typedef __attribute__((ext_vector_type(4)))  float         v4f;

__device__ __forceinline__ v16bf as_bf16x16(v8u u) {
    union { v8u u; v16bf b; } c; c.u = u; return c.b;
}

// pack two f32 -> one dword of two bf16 (lo = first K, hi = second K), RNE
__device__ __forceinline__ unsigned int pk_bf16(float lo, float hi) {
    unsigned int ul = __float_as_uint(lo);
    unsigned int uh = __float_as_uint(hi);
    ul += 0x7FFFu + ((ul >> 16) & 1u);
    uh += 0x7FFFu + ((uh >> 16) & 1u);
    return (ul >> 16) | (uh & 0xFFFF0000u);
}

// ---------------------------------------------------------------------------
// Pack a K x Nc fp32 matrix into the CDNA5 16-bit B-matrix (32x16 per frag)
// VGPR layout.  Fragment f = kb*(Nc/16)+nt ; within a fragment the dword for
// (lane, d) holds elements (k = kb*32 + 8*(lane>>4) + 2*(d&3) + 16*(d>>2) [+1], n = nt*16 +
// (lane&15)).  Stored as out[f*256 + lane*8 + d] so a lane's 8 dwords are a
// contiguous 32B run (two global_load_b128).
// transpose=1 reads X[n*ld + k] (used for Z^T in the decoder).
// ---------------------------------------------------------------------------
__global__ void pack_b_kernel(const float* __restrict__ X,
                              unsigned int* __restrict__ out,
                              int K, int Nc, int ld, int transpose) {
    int tid = blockIdx.x * blockDim.x + threadIdx.x;
    int total = (K >> 5) * (Nc >> 4) * 256;
    if (tid >= total) return;
    int d    = tid & 7;
    int lane = (tid >> 3) & 31;
    int f    = tid >> 8;
    int ntiles = Nc >> 4;
    int nt = f % ntiles;
    int kb = f / ntiles;
    int n  = nt * 16 + (lane & 15);
    int k  = kb * 32 + ((lane >> 4) << 3) + 2 * (d & 3) + ((d >> 2) << 4);
    float x0, x1;
    if (!transpose) {
        x0 = X[(size_t)k * ld + n];
        x1 = X[(size_t)(k + 1) * ld + n];
    } else {
        x0 = X[(size_t)n * ld + k];
        x1 = X[(size_t)n * ld + k + 1];
    }
    out[tid] = pk_bf16(x0, x1);
}

// ---------------------------------------------------------------------------
// Main WMMA GEMM:  C[z] = A[z] (fp32, MxK row-major, lda=K) @ Bfrag (bf16)
// Block = 256 threads = 8 waves.  Block tile: 128 rows x (NT*16) cols.
// Wave w owns rows [blockIdx.y*128 + w*16, +16) x NT*16 cols = NT C-frags.
// A is converted fp32->bf16 in registers per the 16-bit A-fragment layout:
// lane half h = lane>>4 reads row (mBase + lane&15), chunks K+8h..+8h+7 and
// K+16+8h..+23 (four 16B-aligned float4 loads per K-step of 32).
// NT=8 lets layer 2 / decoder stream each A element from HBM exactly once
// (grid.x = 1 for Nc=128) and amortizes the fp32->bf16 conversion over
// 8 WMMAs per K-step.
// ---------------------------------------------------------------------------
template <int NT>
__global__ __launch_bounds__(256) void gemm_wmma_kernel(
    const float* __restrict__ A0, const float* __restrict__ A1,
    const float* __restrict__ A2,
    const unsigned int* __restrict__ Bfrag,
    float* __restrict__ C, long long cStrideZ,
    int K, int Nc, int ldc)
{
    const float* A = (blockIdx.z == 0) ? A0 : (blockIdx.z == 1 ? A1 : A2);
    float* Cp = C + (size_t)blockIdx.z * (size_t)cStrideZ;

    const int lane = threadIdx.x & 31;
    const int wave = threadIdx.x >> 5;
    const int lmod = lane & 15;
    const int h    = lane >> 4;

    const int mBase = blockIdx.y * 128 + wave * 16;
    const int nBase = blockIdx.x * (NT * 16);
    const int ntilesAll = Nc >> 4;

    const float* arow = A + (size_t)(mBase + lmod) * (size_t)K + h * 8;
    const unsigned int* bcol =
        Bfrag + (size_t)(nBase >> 4) * 256 + (size_t)lane * 8;

    v8f acc[NT];
#pragma unroll
    for (int t = 0; t < NT; ++t) acc[t] = (v8f){0, 0, 0, 0, 0, 0, 0, 0};

    const int kblocks = K >> 5;
    for (int kb = 0; kb < kblocks; ++kb) {
        // ---- A fragment: 16 fp32 per lane -> 8 packed bf16 dwords ----
        const float* ap = arow + (size_t)kb * 32;
        v4f a0 = *reinterpret_cast<const v4f*>(ap);          // K = 8h + 0..3
        v4f a1 = *reinterpret_cast<const v4f*>(ap + 4);      // K = 8h + 4..7
        v4f a2 = *reinterpret_cast<const v4f*>(ap + 16);     // K = 16+8h+0..3
        v4f a3 = *reinterpret_cast<const v4f*>(ap + 20);     // K = 16+8h+4..7
        v8u a8;
        a8[0] = pk_bf16(a0[0], a0[1]);
        a8[1] = pk_bf16(a0[2], a0[3]);
        a8[2] = pk_bf16(a1[0], a1[1]);
        a8[3] = pk_bf16(a1[2], a1[3]);
        a8[4] = pk_bf16(a2[0], a2[1]);
        a8[5] = pk_bf16(a2[2], a2[3]);
        a8[6] = pk_bf16(a3[0], a3[1]);
        a8[7] = pk_bf16(a3[2], a3[3]);
        v16bf av = as_bf16x16(a8);

        // ---- B fragments: pre-packed, 2x b128 per fragment ----
        const unsigned int* bp = bcol + (size_t)kb * (size_t)ntilesAll * 256;
#pragma unroll
        for (int t = 0; t < NT; ++t) {
            v8u b;
            *reinterpret_cast<v4u*>(&b) =
                *reinterpret_cast<const v4u*>(bp + t * 256);
            *(reinterpret_cast<v4u*>(&b) + 1) =
                *reinterpret_cast<const v4u*>(bp + t * 256 + 4);
            acc[t] = __builtin_amdgcn_wmma_f32_16x16x32_bf16(
                false, av, false, as_bf16x16(b), (short)0, acc[t],
                false, false);
        }
    }

    // ---- C store: VGPR j holds (m = mBase + 8h + j, n = nBase + 16t + lmod)
#pragma unroll
    for (int j = 0; j < 8; ++j) {
        int m = mBase + 8 * h + j;
        float* crow = Cp + (size_t)m * (size_t)ldc + nBase + lmod;
#pragma unroll
        for (int t = 0; t < NT; ++t) crow[t * 16] = acc[t][j];
    }
}

// ---------------------------------------------------------------------------
// Epilogue: out[i, o] = act( bias[o] + sum_r M_r[i, :] . W_r[o, :] )
// blockDim.x = 128 (= out dim), one block per row i.  FLOP count negligible.
// ---------------------------------------------------------------------------
__global__ __launch_bounds__(128) void combine_kernel(
    const float* __restrict__ M, long long mStrideR,
    const float* __restrict__ Wa, const float* __restrict__ Wb,
    const float* __restrict__ Wc,
    const float* __restrict__ bias,
    float* __restrict__ out, int inDim, int doRelu)
{
    const int i = blockIdx.x;
    const int o = threadIdx.x;
    float s = bias[o];

    const float* m0 = M + (size_t)i * inDim;
    const float* m1 = m0 + (size_t)mStrideR;
    const float* m2 = m1 + (size_t)mStrideR;
    const float* w0 = Wa + (size_t)o * inDim;
    const float* w1 = Wb + (size_t)o * inDim;
    const float* w2 = Wc + (size_t)o * inDim;

    float acc0 = 0.f, acc1 = 0.f, acc2 = 0.f;
    for (int c = 0; c < inDim; ++c) {
        acc0 = fmaf(m0[c], w0[c], acc0);
        acc1 = fmaf(m1[c], w1[c], acc1);
        acc2 = fmaf(m2[c], w2[c], acc2);
    }
    s += acc0 + acc1 + acc2;
    out[(size_t)i * 128 + o] = doRelu ? fmaxf(s, 0.f) : s;
}

// ---------------------------------------------------------------------------
// Host-side orchestration
// ---------------------------------------------------------------------------
extern "C" void kernel_launch(void* const* d_in, const int* in_sizes, int n_in,
                              void* d_out, int out_size, void* d_ws, size_t ws_size,
                              hipStream_t stream) {
    (void)in_sizes; (void)n_in; (void)out_size; (void)ws_size;

    constexpr int N    = 8192;
    constexpr int FEAT = 64;
    constexpr int HID  = 128;
    constexpr int EMB  = 128;

    // setup_inputs() insertion order (W1_i / W2_i interleaved!)
    const float* H    = (const float*)d_in[0];
    const float* Ab   = (const float*)d_in[1];
    const float* Av   = (const float*)d_in[2];
    const float* Ar   = (const float*)d_in[3];
    const float* W1_0 = (const float*)d_in[4];
    const float* W2_0 = (const float*)d_in[5];
    const float* W1_1 = (const float*)d_in[6];
    const float* W2_1 = (const float*)d_in[7];
    const float* W1_2 = (const float*)d_in[8];
    const float* W2_2 = (const float*)d_in[9];
    const float* b1   = (const float*)d_in[10];
    const float* b2   = (const float*)d_in[11];

    float* Z     = (float*)d_out;                       // [N, EMB]
    float* Ahat  = (float*)d_out + (size_t)N * EMB;     // [N, N]

    // workspace layout (byte offsets)
    char* ws = (char*)d_ws;
    unsigned int* Hfrag  = (unsigned int*)(ws + 0);            // 1,048,576 B
    float*        Mws    = (float*)(ws + 1048576);             // 6,291,456 B
    float*        H1     = (float*)(ws + 7340032);             // 4,194,304 B
    unsigned int* H1frag = (unsigned int*)(ws + 11534336);     // 2,097,152 B
    float*        M2ws   = (float*)(ws + 13631488);            // 12,582,912 B
    unsigned int* ZTfrag = (unsigned int*)(ws + 26214400);     // 2,097,152 B

    // ---- Layer 1: M_r = A_r @ H  (K=8192, Nc=64, one A pass) ----
    {
        int total = (N >> 5) * (FEAT >> 4) * 256;              // 262144
        pack_b_kernel<<<total / 256, 256, 0, stream>>>(H, Hfrag, N, FEAT, FEAT, 0);

        dim3 g(FEAT / 64, N / 128, 3);                         // (1, 64, 3)
        gemm_wmma_kernel<4><<<g, 256, 0, stream>>>(
            Ab, Av, Ar, Hfrag, Mws, (long long)N * FEAT, N, FEAT, FEAT);

        // H1 = relu(b1 + sum_r M_r @ W1_r^T)
        combine_kernel<<<N, HID, 0, stream>>>(
            Mws, (long long)N * FEAT, W1_0, W1_1, W1_2, b1, H1, FEAT, 1);
    }

    // ---- Layer 2: M2_r = A_r @ H1  (K=8192, Nc=128, one A pass) ----
    {
        int total = (N >> 5) * (HID >> 4) * 256;               // 524288
        pack_b_kernel<<<total / 256, 256, 0, stream>>>(H1, H1frag, N, HID, HID, 0);

        dim3 g(HID / 128, N / 128, 3);                         // (1, 64, 3)
        gemm_wmma_kernel<8><<<g, 256, 0, stream>>>(
            Ab, Av, Ar, H1frag, M2ws, (long long)N * HID, N, HID, HID);

        // Z = b2 + sum_r M2_r @ W2_r^T   (straight into d_out)
        combine_kernel<<<N, EMB, 0, stream>>>(
            M2ws, (long long)N * HID, W2_0, W2_1, W2_2, b2, Z, HID, 0);
    }

    // ---- Decoder: A_hat = Z @ Z^T  (K=128, Nc=8192) ----
    {
        int total = (EMB >> 5) * (N >> 4) * 256;               // 524288
        pack_b_kernel<<<total / 256, 256, 0, stream>>>(Z, ZTfrag, EMB, N, EMB, 1);

        dim3 g(N / 128, N / 128, 1);                           // (64, 64, 1)
        gemm_wmma_kernel<8><<<g, 256, 0, stream>>>(
            Z, Z, Z, ZTfrag, Ahat, 0LL, EMB, N, N);
    }
}
